// RebornSegmenter_28363964023117
// MI455X (gfx1250) — matrix operand order repaired and checked by reading, not compiled, not run on hardware
//
#include <hip/hip_runtime.h>
#include <hip/hip_bf16.h>

#define Bn 16
#define Tn 8192
#define Cn 256
#define Hn 256

typedef __attribute__((ext_vector_type(16))) _Float16 v16h;
typedef __attribute__((ext_vector_type(8)))  float    v8f;
typedef __attribute__((ext_vector_type(4)))  float    v4f;

union FragH { v16h v; v4f q[2]; };

// ---- workspace layout (bytes) ----
#define O_W1F   0u          // 16 ntile * 56 kstep * 32 lane * 16 halves = 917504 B
#define O_W2F   917504u     // 16 * 24 * 32 * 16 halves                 = 393216 B
#define O_ISB   1310720u    // B*T uchar                                = 131072 B
#define O_SEG   1441792u    // B*T int                                  = 524288 B
#define O_CNT   1966080u    // B*T float                                = 524288 B
#define O_NVAL  2490368u    // B ints

__device__ __forceinline__ v8f zero8() {
    v8f z = {0.f, 0.f, 0.f, 0.f, 0.f, 0.f, 0.f, 0.f};
    return z;
}

__device__ __forceinline__ v8f wmma16(v16h a, v16h b, v8f c) {
    return __builtin_amdgcn_wmma_f32_16x16x32_f16(false, a, false, b, (short)0, c,
                                                  false, false);
}

// A-fragment (16x32 f16) from LDS, row-major ldm=256 halves.
// lane<16: M=lane, K in {0..7}U{16..23}; lane>=16: M=lane-16, K in {8..15}U{24..31}
__device__ __forceinline__ v16h lds_a(const _Float16* base, int row0, int col0) {
    int lane = threadIdx.x & 31;
    int m    = lane & 15;
    int khi  = (lane & 16) >> 1;                 // 0 or 8
    const _Float16* p = base + (row0 + m) * 256 + col0 + khi;
    FragH f;
    f.q[0] = *(const v4f*)(p);                   // K khi..khi+7
    f.q[1] = *(const v4f*)(p + 16);              // K khi+16..khi+23
    return f.v;
}

// B-fragment (32x16 f16) streamed from repacked global buffer: 32 lanes x 16 halves
__device__ __forceinline__ v16h glb_b(const _Float16* wf, int frag) {
    int lane = threadIdx.x & 31;
    const _Float16* p = wf + ((size_t)frag * 32 + lane) * 16;
    FragH f;
    f.q[0] = *(const v4f*)(p);
    f.q[1] = *(const v4f*)(p + 8);
    return f.v;
}

// C/D (16x16 f32) -> relu -> f16 LDS.  VGPR j: lanes0-15 M=j, lanes16-31 M=j+8; N=lane&15
__device__ __forceinline__ void st_cd_relu(_Float16* dst, v8f acc, float bias,
                                           int row0, int col0) {
    int lane = threadIdx.x & 31;
    int col  = col0 + (lane & 15);
    int r    = row0 + ((lane & 16) >> 1);
    #pragma unroll
    for (int j = 0; j < 8; ++j) {
        float v = acc[j] + bias;
        dst[(r + j) * 256 + col] = (_Float16)(v > 0.f ? v : 0.f);
    }
}

// -------- weight repack: fp32 (O,I,W) -> f16 WMMA B-fragments --------
// frag lane L, half h holds Bmat[K = (h&7) + 8*(L>=16) + 16*(h>>3), N = L&15]; GEMM K = tap*256 + c
__global__ void repack_w(const float* __restrict__ w, _Float16* __restrict__ wf,
                         int nk, int taps) {
    int idx   = blockIdx.x * 256 + threadIdx.x;
    int total = 16 * nk * 32 * 16;
    if (idx >= total) return;
    int h     = idx & 15;
    int lane  = (idx >> 4) & 31;
    int kstep = (idx >> 9) % nk;
    int ntile = idx / (512 * nk);
    int kk    = (h & 7) + ((lane >> 4) << 3) + ((h >> 3) << 4);
    int Kg    = kstep * 32 + kk;
    int tap   = Kg >> 8;
    int c     = Kg & 255;
    int n     = ntile * 16 + (lane & 15);
    wf[idx]   = (_Float16)w[(n * Cn + c) * taps + tap];
}

// -------- fused conv1+conv2+conv3+argmax --------
__global__ __launch_bounds__(256) void conv_fused(
    const float* __restrict__ x, const unsigned char* __restrict__ pm,
    const _Float16* __restrict__ w1f, const float* __restrict__ b1,
    const _Float16* __restrict__ w2f, const float* __restrict__ b2,
    const float* __restrict__ w3, const float* __restrict__ b3,
    unsigned char* __restrict__ isb)
{
    __shared__ _Float16 xs[56 * 256];   // x tile [t0-11, t0+43), reused for h2
    __shared__ _Float16 h1s[48 * 256];  // h1 tile [t0-8,  t0+40)

    int b   = blockIdx.y;
    int t0  = blockIdx.x * 32;
    int tid = threadIdx.x;
    int w   = tid >> 5;       // wave id 0..7
    int n0  = w * 2;          // this wave owns N-tiles n0, n0+1

    // stage 0: x fp32 -> f16 LDS (zero-pad outside [0,T))
    for (int r = 0; r < 54; ++r) {
        int t   = t0 - 11 + r;
        float v = (t >= 0 && t < Tn) ? x[((size_t)b * Tn + t) * Cn + tid] : 0.f;
        xs[r * 256 + tid] = (_Float16)v;
    }
    __syncthreads();

    // stage 1: h1 = relu(conv1(x)), M=48 rows [t0-8,t0+40), K=1792
    {
        v8f acc[3][2];
        #pragma unroll
        for (int i = 0; i < 3; ++i)
            for (int j = 0; j < 2; ++j) acc[i][j] = zero8();
        for (int tap = 0; tap < 7; ++tap) {
            #pragma unroll
            for (int cc = 0; cc < 8; ++cc) {
                int ks  = tap * 8 + cc;
                v16h a0 = lds_a(xs,  0 + tap, cc * 32);
                v16h a1 = lds_a(xs, 16 + tap, cc * 32);
                v16h a2 = lds_a(xs, 32 + tap, cc * 32);
                v16h q0 = glb_b(w1f, (n0 + 0) * 56 + ks);
                v16h q1 = glb_b(w1f, (n0 + 1) * 56 + ks);
                acc[0][0] = wmma16(a0, q0, acc[0][0]);
                acc[0][1] = wmma16(a0, q1, acc[0][1]);
                acc[1][0] = wmma16(a1, q0, acc[1][0]);
                acc[1][1] = wmma16(a1, q1, acc[1][1]);
                acc[2][0] = wmma16(a2, q0, acc[2][0]);
                acc[2][1] = wmma16(a2, q1, acc[2][1]);
            }
        }
        float bi0 = b1[(n0 + 0) * 16 + (tid & 15)];
        float bi1 = b1[(n0 + 1) * 16 + (tid & 15)];
        #pragma unroll
        for (int i = 0; i < 3; ++i) {
            st_cd_relu(h1s, acc[i][0], bi0, i * 16, (n0 + 0) * 16);
            st_cd_relu(h1s, acc[i][1], bi1, i * 16, (n0 + 1) * 16);
        }
    }
    __syncthreads();

    // stage 2: h2 = relu(conv2(h1)), M=32 rows [t0,t0+32), K=768; h2 -> xs (reused)
    {
        v8f acc[2][2];
        #pragma unroll
        for (int i = 0; i < 2; ++i)
            for (int j = 0; j < 2; ++j) acc[i][j] = zero8();
        for (int tap = 0; tap < 3; ++tap) {
            #pragma unroll
            for (int cc = 0; cc < 8; ++cc) {
                int ks  = tap * 8 + cc;
                v16h a0 = lds_a(h1s, 7 + tap +  0, cc * 32);
                v16h a1 = lds_a(h1s, 7 + tap + 16, cc * 32);
                v16h q0 = glb_b(w2f, (n0 + 0) * 24 + ks);
                v16h q1 = glb_b(w2f, (n0 + 1) * 24 + ks);
                acc[0][0] = wmma16(a0, q0, acc[0][0]);
                acc[0][1] = wmma16(a0, q1, acc[0][1]);
                acc[1][0] = wmma16(a1, q0, acc[1][0]);
                acc[1][1] = wmma16(a1, q1, acc[1][1]);
            }
        }
        float bi0 = b2[(n0 + 0) * 16 + (tid & 15)];
        float bi1 = b2[(n0 + 1) * 16 + (tid & 15)];
        #pragma unroll
        for (int i = 0; i < 2; ++i) {
            st_cd_relu(xs, acc[i][0], bi0, i * 16, (n0 + 0) * 16);
            st_cd_relu(xs, acc[i][1], bi1, i * 16, (n0 + 1) * 16);
        }
    }
    __syncthreads();

    // stage 3: 2-channel projection + argmax -> boundary byte
    if (tid < 32) {
        int t = t0 + tid;
        float l0 = b3[0], l1 = b3[1];
        for (int c = 0; c < 256; ++c) {
            float hv = (float)xs[tid * 256 + c];
            l0 += hv * w3[c];
            l1 += hv * w3[256 + c];
        }
        int pad = pm[(size_t)b * Tn + t];
        isb[(size_t)b * Tn + t] = (l1 > l0 && !pad) ? 1 : 0;
    }
}

// -------- per-batch inclusive scan of boundaries -> seg ids --------
__global__ __launch_bounds__(256) void scan_kernel(
    const unsigned char* __restrict__ isb, const unsigned char* __restrict__ pm,
    int* __restrict__ seg, int* __restrict__ nval)
{
    __shared__ int psum[257];
    __shared__ int anyv;
    int b    = blockIdx.x;
    int tid  = threadIdx.x;
    const int CH = Tn / 256;              // 32
    size_t base  = (size_t)b * Tn + (size_t)tid * CH;
    int s = 0, av = 0;
    for (int i = 0; i < CH; ++i) { s += isb[base + i]; av |= !pm[base + i]; }
    psum[tid + 1] = s;
    if (tid == 0) { psum[0] = 0; anyv = 0; }
    __syncthreads();
    if (av) atomicOr(&anyv, 1);
    if (tid == 0) for (int i = 1; i <= 256; ++i) psum[i] += psum[i - 1];
    __syncthreads();
    int run = psum[tid];
    for (int i = 0; i < CH; ++i) {
        run += isb[base + i];
        seg[base + i] = pm[base + i] ? Tn : run;
    }
    if (tid == 0) nval[b] = anyv ? psum[256] + 1 : 0;
}

// -------- zero accumulators --------
__global__ void zero_kernel(float* __restrict__ out, float* __restrict__ cnt) {
    size_t n1 = (size_t)Bn * Tn * Cn + (size_t)Bn * Tn;
    size_t n2 = (size_t)Bn * Tn;
    size_t i  = (size_t)blockIdx.x * blockDim.x + threadIdx.x;
    size_t st = (size_t)gridDim.x * blockDim.x;
    for (size_t k = i; k < n1; k += st) out[k] = 0.f;
    for (size_t k = i; k < n2; k += st) cnt[k] = 0.f;
}

// -------- segment pooling: run-merged f32 atomics --------
__global__ __launch_bounds__(256) void pool_kernel(
    const float* __restrict__ x, const int* __restrict__ seg,
    float* __restrict__ outsum, float* __restrict__ cnt)
{
    int b  = blockIdx.y;
    int t0 = blockIdx.x * 32;
    int c  = threadIdx.x;
    float acc = 0.f, cacc = 0.f;
    int prev = -1;
    for (int i = 0; i < 32; ++i) {
        int t = t0 + i;
        int s = seg[(size_t)b * Tn + t];
        if (s != prev) {
            if (prev >= 0 && prev < Tn) {
                unsafeAtomicAdd(&outsum[((size_t)b * Tn + prev) * Cn + c], acc);
                if (c == 0) unsafeAtomicAdd(&cnt[(size_t)b * Tn + prev], cacc);
            }
            acc = 0.f; cacc = 0.f; prev = s;
        }
        acc  += x[((size_t)b * Tn + t) * Cn + c];
        cacc += 1.f;
    }
    if (prev >= 0 && prev < Tn) {
        unsafeAtomicAdd(&outsum[((size_t)b * Tn + prev) * Cn + c], acc);
        if (c == 0) unsafeAtomicAdd(&cnt[(size_t)b * Tn + prev], cacc);
    }
}

// -------- mean + new_pad --------
__global__ void finalize_kernel(float* __restrict__ out, const float* __restrict__ cnt,
                                const int* __restrict__ nval) {
    size_t i    = (size_t)blockIdx.x * 256 + threadIdx.x;
    size_t nlog = (size_t)Bn * Tn * Cn;
    if (i < nlog) {
        int b = (int)(i / ((size_t)Tn * Cn));
        int s = (int)((i / Cn) % Tn);
        float cv = cnt[(size_t)b * Tn + s];
        out[i] = (cv > 0.5f) ? out[i] / cv : 0.f;
    } else if (i < nlog + (size_t)Bn * Tn) {
        size_t j = i - nlog;
        int b = (int)(j / Tn), s = (int)(j % Tn);
        out[i] = (s >= nval[b]) ? 1.f : 0.f;
    }
}

extern "C" void kernel_launch(void* const* d_in, const int* in_sizes, int n_in,
                              void* d_out, int out_size, void* d_ws, size_t ws_size,
                              hipStream_t stream) {
    const float* x          = (const float*)d_in[0];
    const unsigned char* pm = (const unsigned char*)d_in[1];  // jnp bool = 1 byte
    const float* w1 = (const float*)d_in[2];
    const float* b1 = (const float*)d_in[3];
    const float* w2 = (const float*)d_in[4];
    const float* b2 = (const float*)d_in[5];
    const float* w3 = (const float*)d_in[6];
    const float* b3 = (const float*)d_in[7];

    char* ws            = (char*)d_ws;
    _Float16* w1f       = (_Float16*)(ws + O_W1F);
    _Float16* w2f       = (_Float16*)(ws + O_W2F);
    unsigned char* isb  = (unsigned char*)(ws + O_ISB);
    int* seg            = (int*)(ws + O_SEG);
    float* cnt          = (float*)(ws + O_CNT);
    int* nval           = (int*)(ws + O_NVAL);
    float* out          = (float*)d_out;

    repack_w<<<(16 * 56 * 32 * 16 + 255) / 256, 256, 0, stream>>>(w1, w1f, 56, 7);
    repack_w<<<(16 * 24 * 32 * 16 + 255) / 256, 256, 0, stream>>>(w2, w2f, 24, 3);
    zero_kernel<<<2048, 256, 0, stream>>>(out, cnt);
    conv_fused<<<dim3(Tn / 32, Bn), 256, 0, stream>>>(x, pm, w1f, b1, w2f, b2,
                                                      w3, b3, isb);
    scan_kernel<<<Bn, 256, 0, stream>>>(isb, pm, seg, nval);
    pool_kernel<<<dim3(Tn / 32, Bn), 256, 0, stream>>>(x, seg, out, cnt);
    size_t ntot = (size_t)Bn * Tn * Cn + (size_t)Bn * Tn;
    finalize_kernel<<<(unsigned)((ntot + 255) / 256), 256, 0, stream>>>(out, cnt, nval);
}
